// DenseRandlanetKernel_71717363908926
// MI455X (gfx1250) — matrix operand order repaired and checked by reading, not compiled, not run on hardware
//
#include <hip/hip_runtime.h>

// Fused RandLA-Net attention block for MI455X (gfx1250, wave32, bf16 WMMA).
// All LDS operands are stored pre-swizzled in WMMA fragment layout so each
// lane loads its 16 bf16 operand elements as one 32-byte vector (ds_load_b128
// pairs) instead of 16 scalar ds_load_u16.
//
// One wave handles 16 points: per point it builds enc (16x10), runs
// rel_point_pos_nn + the 16x128x128 attention GEMM on
// v_wmma_f32_16x16x32_bf16, does a cross-half-wave softmax over the 16
// neighbors, accumulates agg rows, then one batched M=16 WMMA pass applies
// global_nn for its 16 points.

#define WAVES_PER_WG 8

typedef __attribute__((ext_vector_type(16))) __bf16 v16bf;
typedef __attribute__((ext_vector_type(2)))  __bf16 v2bf;
typedef __attribute__((ext_vector_type(8)))  float  v8f;
typedef __attribute__((ext_vector_type(4)))  float  v4f;

namespace {
constexpr int kB    = 4;
constexpr int kN    = 16384;
constexpr int kK    = 16;      // neighbors
constexpr int kCIN  = 64;
constexpr int kDREL = 64;
constexpr int kCMID = 128;
constexpr int kCOUT = 128;
constexpr int kPtsPerWave = 16;

// One fragment chunk = 32 lanes x 16 bf16 elements = 512 bf16 (1 KB).
constexpr int kChunk          = 32 * 16;
constexpr int kWrelFragElems  = 4  * kChunk;   // 4 N-tiles, single K-chunk
constexpr int kWattFragElems  = 32 * kChunk;   // 4 K-chunks x 8 N-tiles
constexpr int kWglobFragElems = 32 * kChunk;
constexpr int kFFragElems     = 4  * kChunk;   // f[16x128] in A-frag layout
constexpr int kAggFragElems   = 4  * kChunk;   // agg[16x128] in A-frag layout
constexpr int kSmemElems      = kWrelFragElems + kWattFragElems + kWglobFragElems +
                                WAVES_PER_WG * (kFFragElems + kAggFragElems);
}

__device__ __forceinline__ __bf16 f2bf(float x) { return (__bf16)x; }

__device__ __forceinline__ void lds_fence() {
  // DS ops from one wave complete in order; fence the compiler and wait so
  // cross-lane LDS writes are retired before the reads below.
  asm volatile("s_wait_dscnt 0x0" ::: "memory");
}

// ---- fragment layout maps (wave32 16x16x32 bf16 WMMA) ----------------------
// A (16x32): lane l (hfA = l>=16) element e <-> K = (e<8 ? e : e+8) + (hfA?8:0)
//   inverse: given K: hfA = (K&8)!=0, e = ((K&16)?8:0) + (K&7), lane = M + hfA*16
// B (32x16): lane l (hfB = l>=16) element e <-> K = (e<8 ? e : e+8) + (hfB?8:0), N = l&15
//   inverse: given (K,N): lane = N + ((K&8)?16:0), e = ((K&16)?8:0) + (K&7)

__global__ void randlanet_fused(const float* __restrict__ x,      // [B,CIN,N]
                                const float* __restrict__ pos,    // [B,N,3]
                                const int*   __restrict__ nidx,   // [B,N,K]
                                const float* __restrict__ Wrel,   // [10,DREL]
                                const float* __restrict__ brel,   // [DREL]
                                const float* __restrict__ Watt,   // [CMID,CMID]
                                const float* __restrict__ Wglob,  // [CMID,COUT]
                                const float* __restrict__ bglob,  // [COUT]
                                float* __restrict__ out)          // [B,COUT,N]
{
  extern __shared__ __align__(16) char smem_raw[];
  __bf16* sWrel  = (__bf16*)smem_raw;
  __bf16* sWatt  = sWrel  + kWrelFragElems;
  __bf16* sWglob = sWatt  + kWattFragElems;
  __bf16* sFrags = sWglob + kWglobFragElems;

  const int tid  = threadIdx.x;
  const int wave = tid >> 5;
  const int lane = tid & 31;
  const int row  = lane & 15;           // A row / B column within a tile
  const bool hi  = lane >= 16;

  __bf16* sF   = sFrags + wave * (kFFragElems + kAggFragElems);
  __bf16* sAgg = sF + kFFragElems;

  // ---- cooperative weight staging: fp32 -> bf16 B-fragment layout ---------
  // W_att / W_glob [128 in][128 out]: pack consecutive-K pairs into b32 stores.
  for (int i = tid; i < kCMID * kCMID / 2; i += blockDim.x) {
    const int k2 = (i >> 7) << 1;                // even input channel
    const int o  = i & 127;                      // output channel
    v2bf pa, pg;
    pa[0] = f2bf(Watt [(size_t)k2       * kCMID + o]);
    pa[1] = f2bf(Watt [(size_t)(k2 + 1) * kCMID + o]);
    pg[0] = f2bf(Wglob[(size_t)k2       * kCOUT + o]);
    pg[1] = f2bf(Wglob[(size_t)(k2 + 1) * kCOUT + o]);
    const int kt = k2 >> 5, K = k2 & 31;
    const int idx = (((kt << 3) + (o >> 4)) * 32 + ((o & 15) + (((K >> 3) & 1) << 4))) * 16 +
                    (((K >> 4) << 3) + (K & 7));
    *(v2bf*)(sWatt  + idx) = pa;
    *(v2bf*)(sWglob + idx) = pg;
  }
  // W_rel [10 in][64 out], K padded 10 -> 32 with zeros; single K-chunk.
  for (int i = tid; i < 16 * kDREL; i += blockDim.x) {
    const int k2 = (i >> 6) << 1;
    const int o  = i & 63;
    v2bf p;
    p[0] = f2bf(k2     < 10 ? Wrel[(size_t)k2       * kDREL + o] : 0.0f);
    p[1] = f2bf(k2 + 1 < 10 ? Wrel[(size_t)(k2 + 1) * kDREL + o] : 0.0f);
    const int idx = ((o >> 4) * 32 + ((o & 15) + (((k2 >> 3) & 1) << 4))) * 16 + (((k2 >> 4) << 3) + (k2 & 7));
    *(v2bf*)(sWrel + idx) = p;
  }
  __syncthreads();

  const int gbase = (blockIdx.x * WAVES_PER_WG + wave) * kPtsPerWave;
  const int b     = gbase >> 14;        // kN == 16384; groups never cross b
  const int nbase = gbase & (kN - 1);
  const float* xb = x + (size_t)b * kCIN * kN;

  // ======================= per-point attention phase =======================
  for (int p = 0; p < kPtsPerWave; ++p) {
    const int n = nbase + p;

    // Both half-waves handle neighbor `row` (rows 0..15 of the A matrix).
    const int j = nidx[((size_t)b * kN + n) * kK + row];
    const float cx = pos[((size_t)b * kN + n) * 3 + 0];
    const float cy = pos[((size_t)b * kN + n) * 3 + 1];
    const float cz = pos[((size_t)b * kN + n) * 3 + 2];
    const float px = pos[((size_t)b * kN + j) * 3 + 0];
    const float py = pos[((size_t)b * kN + j) * 3 + 1];
    const float pz = pos[((size_t)b * kN + j) * 3 + 2];
    const float rx = px - cx, ry = py - cy, rz = pz - cz;
    const float dist = sqrtf(rx * rx + ry * ry + rz * rz);
    float enc[10] = {cx, cy, cz, px, py, pz, rx, ry, rz, dist};

    // A-operand: enc 16x32 bf16 (K zero-padded). Lanes 0-15 hold K={0..7,16..23},
    // lanes 16-31 K={8..15,24..31} -> only elements {0..7} / {0,1} are nonzero.
    v16bf aEnc;
#pragma unroll
    for (int e = 0; e < 16; ++e) aEnc[e] = f2bf(0.0f);
    if (!hi) {
#pragma unroll
      for (int e = 0; e < 8; ++e) aEnc[e] = f2bf(enc[e]);
    } else {
      aEnc[0] = f2bf(enc[8]);
      aEnc[1] = f2bf(enc[9]);
    }

    // rel_point_pos_nn: rp[16][64] = relu(enc @ Wrel + brel), scatter into the
    // A-fragment image of f (channels 0..63 = K-chunks 0..1).
#pragma unroll
    for (int t = 0; t < 4; ++t) {
      const v16bf bW = *(const v16bf*)(sWrel + (t * 32 + lane) * 16);
      v8f acc = {};
      acc = __builtin_amdgcn_wmma_f32_16x16x32_bf16(
          false, aEnc, false, bW, (short)0, acc, false, false);
      const float bias = brel[t * 16 + row];
      const int e  = ((t & 1) ? 8 : 0) + (row & 7);       // c = t*16+row
      const int hf = (row & 8) ? 16 : 0;
#pragma unroll
      for (int v = 0; v < 8; ++v) {
        const int m = v + (hi ? 8 : 0);
        sF[(t >> 1) * kChunk + (m + hf) * 16 + e] =
            f2bf(fmaxf(acc[v] + bias, 0.0f));
      }
    }

    // Gathered neighbor features -> f channels 64..127 (K-chunks 2..3),
    // written directly as one full 16-element A-fragment per lane.
#pragma unroll
    for (int kt = 2; kt < 4; ++kt) {
      v16bf tmp;
#pragma unroll
      for (int e = 0; e < 16; ++e) {
        const int K  = (e < 8 ? e : e + 8) + (hi ? 8 : 0);
        const int ch = kt * 32 + K - kDREL;               // x input channel
        tmp[e] = f2bf(xb[(size_t)ch * kN + j]);
      }
      *(v16bf*)(sF + kt * kChunk + lane * 16) = tmp;
    }
    lds_fence();

    // Hoisted A-fragments of f (reused across all 8 output tiles).
    v16bf aF[4];
#pragma unroll
    for (int kt = 0; kt < 4; ++kt)
      aF[kt] = *(const v16bf*)(sF + kt * kChunk + lane * 16);

    // attention GEMM + softmax(K) + weighted sum, tile by tile (channels).
#pragma unroll
    for (int t2 = 0; t2 < 8; ++t2) {
      v8f acc = {};
#pragma unroll
      for (int kt = 0; kt < 4; ++kt) {
        const v16bf bA =
            *(const v16bf*)(sWatt + (((kt << 3) + t2) * 32 + lane) * 16);
        acc = __builtin_amdgcn_wmma_f32_16x16x32_bf16(
            false, aF[kt], false, bA, (short)0, acc, false, false);
      }

      // softmax over the 16 neighbors (M axis = 8 VGPRs x 2 half-waves)
      float m = acc[0];
#pragma unroll
      for (int v = 1; v < 8; ++v) m = fmaxf(m, acc[v]);
      m = fmaxf(m, __shfl_xor(m, 16, 32));
      float ex[8], s = 0.0f;
#pragma unroll
      for (int v = 0; v < 8; ++v) { ex[v] = __expf(acc[v] - m); s += ex[v]; }
      s += __shfl_xor(s, 16, 32);
      const float inv = 1.0f / s;

      // agg[c] = sum_k att[k][c] * f[k][c]; re-read f in C/D orientation from
      // the A-fragment image (transpose access -> scalar b16 loads).
      const int eRe = ((t2 & 1) ? 8 : 0) + (row & 7);
      const int hf  = (row & 8) ? 16 : 0;
      const int ktc = t2 >> 1;
      float aggp = 0.0f;
#pragma unroll
      for (int v = 0; v < 8; ++v) {
        const int M = v + (hi ? 8 : 0);
        const float fv = (float)sF[ktc * kChunk + (M + hf) * 16 + eRe];
        aggp += ex[v] * inv * fv;
      }
      aggp += __shfl_xor(aggp, 16, 32);
      if (!hi) sAgg[ktc * kChunk + (p + hf) * 16 + eRe] = f2bf(aggp);
    }
  }
  lds_fence();

  // ================= global_nn: batched M=16 WMMA over 16 points ===========
  v16bf aA[4];
#pragma unroll
  for (int kt = 0; kt < 4; ++kt)
    aA[kt] = *(const v16bf*)(sAgg + kt * kChunk + lane * 16);

#pragma unroll
  for (int t = 0; t < 8; ++t) {
    v8f acc = {};
#pragma unroll
    for (int kt = 0; kt < 4; ++kt) {
      const v16bf bG =
          *(const v16bf*)(sWglob + (((kt << 3) + t) * 32 + lane) * 16);
      acc = __builtin_amdgcn_wmma_f32_16x16x32_bf16(
          false, aA[kt], false, bG, (short)0, acc, false, false);
    }
    const int c = t * 16 + row;
    const float bias = bglob[c];
    // rows of the D tile are consecutive point ids -> vectorized b128 stores
    v4f o0, o1;
#pragma unroll
    for (int v = 0; v < 4; ++v) o0[v] = fmaxf(acc[v] + bias, 0.0f);
#pragma unroll
    for (int v = 0; v < 4; ++v) o1[v] = fmaxf(acc[v + 4] + bias, 0.0f);
    float* dst = out + ((size_t)b * kCOUT + c) * kN + nbase + (hi ? 8 : 0);
    *(v4f*)(dst + 0) = o0;
    *(v4f*)(dst + 4) = o1;
  }
}

extern "C" void kernel_launch(void* const* d_in, const int* in_sizes, int n_in,
                              void* d_out, int out_size, void* d_ws, size_t ws_size,
                              hipStream_t stream) {
  const float* x     = (const float*)d_in[0];
  const float* pos   = (const float*)d_in[1];
  const int*   nidx  = (const int*)  d_in[2];
  const float* Wrel  = (const float*)d_in[3];
  const float* brel  = (const float*)d_in[4];
  const float* Watt  = (const float*)d_in[5];
  const float* Wglob = (const float*)d_in[6];
  const float* bglob = (const float*)d_in[7];
  float* out = (float*)d_out;

  const int totalPoints = kB * kN;                          // 65536
  const int ptsPerWG    = WAVES_PER_WG * kPtsPerWave;       // 128
  dim3 grid(totalPoints / ptsPerWG);                        // 512
  dim3 block(WAVES_PER_WG * 32);                            // 256 (8 waves)
  size_t smemBytes = (size_t)kSmemElems * 2;                // bf16 elements

  randlanet_fused<<<grid, block, smemBytes, stream>>>(
      x, pos, nidx, Wrel, brel, Watt, Wglob, bglob, out);
}
